// LEM_36601711297045
// MI455X (gfx1250) — matrix operand order repaired and checked by reading, not compiled, not run on hardware
//
#include <hip/hip_runtime.h>
#include <hip/hip_bf16.h>

typedef __attribute__((ext_vector_type(16))) __bf16 v16bf;
typedef __attribute__((ext_vector_type(8)))  float  v8f;

#define DD   768
#define CC   512
#define SS   512
#define BBATCH 64
#define NROWS (BBATCH * SS)   /* 32768 token rows */
#define EPSN 1e-8f

__device__ __forceinline__ float wave_sum32(float v) {
    v += __shfl_xor(v, 1, 32);
    v += __shfl_xor(v, 2, 32);
    v += __shfl_xor(v, 4, 32);
    v += __shfl_xor(v, 8, 32);
    v += __shfl_xor(v, 16, 32);
    return v;
}

// ---------------------------------------------------------------------------
// K1: normalize label embeddings, split into bf16 hi/lo, store [C][D] row-major
// (row-major Ln == column-major B matrix; each B-frag lane reads 16 contiguous K of its column)
// one wave per class row; grid 64 x 256 = 512 waves = C rows
// ---------------------------------------------------------------------------
__global__ __launch_bounds__(256) void k_label_prep(
    const float* __restrict__ L, __bf16* __restrict__ Lhi, __bf16* __restrict__ Llo)
{
    int wid  = (blockIdx.x * blockDim.x + threadIdx.x) >> 5;   // class row
    int lane = threadIdx.x & 31;
    const float* row = L + (size_t)wid * DD;
    float ss = 0.f;
    #pragma unroll
    for (int j = 0; j < DD / 32; ++j) { float x = row[lane + 32 * j]; ss += x * x; }
    ss = wave_sum32(ss);
    float scale = 1.f / fmaxf(sqrtf(ss), EPSN);
    #pragma unroll
    for (int j = 0; j < DD / 32; ++j) {
        int idx = lane + 32 * j;
        float x = row[idx] * scale;
        __bf16 h = (__bf16)x;
        __bf16 l = (__bf16)(x - (float)h);
        Lhi[(size_t)wid * DD + idx] = h;
        Llo[(size_t)wid * DD + idx] = l;
    }
}

// ---------------------------------------------------------------------------
// K2: per-token 1/max(||V[row]||, eps).  One wave per row.
// ---------------------------------------------------------------------------
__global__ __launch_bounds__(256) void k_invnorm(
    const float* __restrict__ V, float* __restrict__ inv)
{
    int wid  = (blockIdx.x * blockDim.x + threadIdx.x) >> 5;
    int lane = threadIdx.x & 31;
    const float* row = V + (size_t)wid * DD;
    float ss = 0.f;
    #pragma unroll
    for (int j = 0; j < DD / 32; ++j) { float x = row[lane + 32 * j]; ss += x * x; }
    ss = wave_sum32(ss);
    if (lane == 0) inv[wid] = 1.f / fmaxf(sqrtf(ss), EPSN);
}

// ---------------------------------------------------------------------------
// K3: fused (Vn @ Ln^T) + max over classes, hi/lo split bf16 WMMA.
// One wave owns a 16-row token tile and sweeps all 512 classes.
// 4 N-groups x 24 K-tiles x 8 N-tiles x 3 wmma = 2304 v_wmma per wave.
// grid 256 x block 256 = 2048 waves = 32768 rows. EXEC all-ones throughout.
// ---------------------------------------------------------------------------
__global__ __launch_bounds__(256) void k_sims_max(
    const float* __restrict__ V,
    const __bf16* __restrict__ Lhi, const __bf16* __restrict__ Llo,
    const float* __restrict__ inv, float* __restrict__ m)
{
    int wid   = (blockIdx.x * blockDim.x + threadIdx.x) >> 5;
    int lane  = threadIdx.x & 31;
    int mbase = wid * 16;
    int rowm  = lane & 15;         // M row within tile (lane halves share rows)
    int half  = lane >> 4;         // 0: K-lo half, 1: K-hi half of frag

    const float* Vrow = V + (size_t)(mbase + rowm) * DD;
    float is = inv[mbase + rowm];

    v8f rmax;
    #pragma unroll
    for (int r = 0; r < 8; ++r) rmax[r] = -3.0e38f;

    for (int g = 0; g < 4; ++g) {               // N-groups of 8 tiles (128 classes)
        v8f acc[8];
        #pragma unroll
        for (int n = 0; n < 8; ++n)
            #pragma unroll
            for (int r = 0; r < 8; ++r) acc[n][r] = 0.f;

        for (int kt = 0; kt < DD / 32; ++kt) {  // 24 K-tiles of 32
            // ---- build A frags (16x32 bf16 layout) from fp32 V, scaled + split
            int c0 = kt * 32 + half * 16;       // lane<16: K 0..7 ; lane>=16: K 16..23
            float4 q[4];
            q[0] = ((const float4*)(Vrow + c0))[0];
            q[1] = ((const float4*)(Vrow + c0))[1];
            q[2] = ((const float4*)(Vrow + c0 + 8))[0];    // +8: K 8..15 / 24..31
            q[3] = ((const float4*)(Vrow + c0 + 8))[1];
            const float* xs = (const float*)q;
            v16bf ahi, alo;
            #pragma unroll
            for (int i = 0; i < 16; ++i) {
                float x = xs[i] * is;
                __bf16 h = (__bf16)x;
                ahi[i] = h;
                alo[i] = (__bf16)(x - (float)h);
            }

            int kb = kt * 32 + half * 16;       // B frag: 16 contiguous K per lane
            #pragma unroll
            for (int n = 0; n < 8; ++n) {
                int col = (g * 8 + n) * 16 + rowm;     // class index = B column
                v16bf bh = *(const v16bf*)(Lhi + (size_t)col * DD + kb);
                v16bf bl = *(const v16bf*)(Llo + (size_t)col * DD + kb);
                acc[n] = __builtin_amdgcn_wmma_f32_16x16x32_bf16(
                             false, ahi, false, bh, (short)0, acc[n], false, false);
                acc[n] = __builtin_amdgcn_wmma_f32_16x16x32_bf16(
                             false, ahi, false, bl, (short)0, acc[n], false, false);
                acc[n] = __builtin_amdgcn_wmma_f32_16x16x32_bf16(
                             false, alo, false, bh, (short)0, acc[n], false, false);
            }
        }
        #pragma unroll
        for (int n = 0; n < 8; ++n)
            #pragma unroll
            for (int r = 0; r < 8; ++r) rmax[r] = fmaxf(rmax[r], acc[n][r]);
    }

    // per-row max across the 16 lanes of each half (columns of the C/D tile)
    float o[8];
    #pragma unroll
    for (int r = 0; r < 8; ++r) {
        float v = rmax[r];
        v = fmaxf(v, __shfl_xor(v, 1, 32));
        v = fmaxf(v, __shfl_xor(v, 2, 32));
        v = fmaxf(v, __shfl_xor(v, 4, 32));
        v = fmaxf(v, __shfl_xor(v, 8, 32));
        o[r] = v;
    }
    if (rowm == 0) {                 // lanes 0 (rows 0..7) and 16 (rows 8..15)
        int rb = mbase + half * 8;
        #pragma unroll
        for (int r = 0; r < 8; ++r) m[rb + r] = o[r];
    }
}

// ---------------------------------------------------------------------------
// K4: softmax over sequence dim per batch. one block per b.
// ---------------------------------------------------------------------------
__global__ __launch_bounds__(256) void k_softmax(
    const float* __restrict__ m, float* __restrict__ beta)
{
    int b = blockIdx.x, tid = threadIdx.x;
    __shared__ float red[256];
    float a0 = m[b * SS + tid], a1 = m[b * SS + tid + 256];
    red[tid] = fmaxf(a0, a1);
    __syncthreads();
    for (int off = 128; off > 0; off >>= 1) {
        if (tid < off) red[tid] = fmaxf(red[tid], red[tid + off]);
        __syncthreads();
    }
    float mx = red[0];
    __syncthreads();
    float e0 = __expf(a0 - mx), e1 = __expf(a1 - mx);
    red[tid] = e0 + e1;
    __syncthreads();
    for (int off = 128; off > 0; off >>= 1) {
        if (tid < off) red[tid] += red[tid + off];
        __syncthreads();
    }
    float s = red[0];
    beta[b * SS + tid]       = e0 / s;
    beta[b * SS + tid + 256] = e1 / s;
}

// ---------------------------------------------------------------------------
// K5: z[b,:] = sum_s beta[b,s] * V[b,s,:].  one block per b, 3 cols/thread.
// ---------------------------------------------------------------------------
__global__ __launch_bounds__(256) void k_z(
    const float* __restrict__ V, const float* __restrict__ beta, float* __restrict__ z)
{
    int b = blockIdx.x, tid = threadIdx.x;
    const float* Vb = V + (size_t)b * SS * DD;
    const float* bb = beta + b * SS;
    float a0 = 0.f, a1 = 0.f, a2 = 0.f;
    for (int s = 0; s < SS; ++s) {
        float w = bb[s];
        const float* r = Vb + (size_t)s * DD;
        a0 += w * r[tid];
        a1 += w * r[tid + 256];
        a2 += w * r[tid + 512];
    }
    z[b * DD + tid]       = a0;
    z[b * DD + tid + 256] = a1;
    z[b * DD + tid + 512] = a2;
}

// ---------------------------------------------------------------------------
// K6: out[b,c] = dot(z[b,:], fc_w[c,:]) + fc_b[c].  one block per b.
// ---------------------------------------------------------------------------
__global__ __launch_bounds__(256) void k_fc(
    const float* __restrict__ z, const float* __restrict__ W,
    const float* __restrict__ bias, float* __restrict__ out)
{
    int b = blockIdx.x, tid = threadIdx.x;
    __shared__ float zs[DD];
    for (int i = tid; i < DD; i += 256) zs[i] = z[b * DD + i];
    __syncthreads();
    for (int c = tid; c < CC; c += 256) {
        const float* wr = W + (size_t)c * DD;
        float acc = bias[c];
        for (int d = 0; d < DD; d += 4) {
            float4 w4 = *(const float4*)(wr + d);
            acc += zs[d] * w4.x + zs[d + 1] * w4.y + zs[d + 2] * w4.z + zs[d + 3] * w4.w;
        }
        out[b * CC + c] = acc;
    }
}

extern "C" void kernel_launch(void* const* d_in, const int* in_sizes, int n_in,
                              void* d_out, int out_size, void* d_ws, size_t ws_size,
                              hipStream_t stream) {
    const float* V   = (const float*)d_in[0];   // [64,512,768]
    const float* L   = (const float*)d_in[1];   // [512,768]
    const float* fcw = (const float*)d_in[2];   // [512,768]
    const float* fcb = (const float*)d_in[3];   // [512]

    float* out = (float*)d_out;                 // [64,512]
    float* z   = out + BBATCH * CC;             // [64,768] (second tuple element)

    // workspace layout (~2 MB, 256B-aligned slices)
    char* w = (char*)d_ws;
    __bf16* Lhi  = (__bf16*)(w);                         //  768 KB
    __bf16* Llo  = (__bf16*)(w + 786432);                //  768 KB
    float*  inv  = (float*)(w + 1572864);                //  128 KB
    float*  m    = (float*)(w + 1703936);                //  128 KB
    float*  beta = (float*)(w + 1835008);                //  128 KB

    k_label_prep<<<CC / 8,    256, 0, stream>>>(L, Lhi, Llo);
    k_invnorm   <<<NROWS / 8, 256, 0, stream>>>(V, inv);
    k_sims_max  <<<NROWS / 128, 256, 0, stream>>>(V, Lhi, Llo, inv, m);
    k_softmax   <<<BBATCH,    256, 0, stream>>>(m, beta);
    k_z         <<<BBATCH,    256, 0, stream>>>(V, beta, z);
    k_fc        <<<BBATCH,    256, 0, stream>>>(z, fcw, fcb, out);
}
